// recon_net_3350074491393
// MI455X (gfx1250) — matrix operand order repaired and checked by reading, not compiled, not run on hardware
//
#include <hip/hip_runtime.h>
#include <hip/hip_bf16.h>

// ---------------------------------------------------------------------------
// MI455X (gfx1250, wave32) implementation.
//
// Pipeline (all on `stream`, deterministic):
//   K0  cvt_swz_kernel      : Wc,W1c,Wf,W1f f32 -> f16, pre-swizzled into
//                             WMMA B-fragment-major order (one-time, ws)
//   K1  d2min_kernel        : per detect point, min sqdist to 8192 pc points
//                             (pc tiles staged to LDS by the TDM)
//   K2  partition_kernel    : stable rank of d2min -> deterministic
//                             close/far compaction (matches stable argsort)
//   K3  knn_mlp_kernel<K>   : 16 queries per workgroup:
//                             - TDM-staged pc tiles, per-thread top-K in
//                               registers (unrolled bubble insert)
//                             - 16-way LDS merge -> exact top-K per query
//                             - IDW weights, coalesced feature gather
//                             - MLP via v_wmma_f32_16x16x32_f16 with
//                               A frags = 2x ds_load_b128 (swizzled LDS),
//                               B frags = 2x global_load_b128 (swizzled, L2)
//
// WMMA fragment layouts (wave32, CDNA5 ISA):
//   A 16x32 f16 : lane L<16 holds row L, K=0..15; lane L+16 holds
//                 row L, K=16..31 (k = 16*h + e, see a_slot)
//   B 32x16 f16 : lane L holds col (L&15), K = 16*(L>>4)+e (same contiguous order as A here)
//   C/D 16x16 f32: elem j of lane L -> row j+8*(L>>4), col (L&15)
// Swizzled storage: fragment-major, offset = ((tile*32)+lane)*16 + e halfs.
// ---------------------------------------------------------------------------

typedef __attribute__((ext_vector_type(16))) _Float16     v16h;
typedef __attribute__((ext_vector_type(8)))  float        v8f;
typedef __attribute__((ext_vector_type(4)))  unsigned int v4u;
typedef __attribute__((ext_vector_type(8)))  int          v8i;
typedef __attribute__((ext_vector_type(4)))  int          v4i;

union F16x16 { v16h v; _Float16 e[16]; v4u u[2]; };
union F32x8  { v8f  v; float    e[8];  };

#define NB     2
#define NPC    8192
#define NDET   12288
#define NCH    256
#define MCLOSE 8192            /* N*2/3 */
#define NFAR   4096

// Map a K-column index (0..31 within a 32-chunk) to the A-fragment
// (lane-half h, element e) slot:  k = 16*h + e.
__device__ __forceinline__ void a_slot(int rem, int& h, int& e) {
  h = (rem >> 4) & 1;
  e = rem & 15;
}

// ---- Tensor Data Mover: 1-D DMA of n_dwords from global to LDS -------------
// D# per CDNA5 ISA. Issue from ONE wave, then
// s_wait_tensorcnt(0) before the workgroup barrier publishes the tile.
// LDS byte address = low 32 bits of the shared-aperture flat pointer
// This toolchain exposes the 6-arg builtin:
//   (uint32x4 g0, int32x8 g1, int32x4 g2, int32x4 g3, int32x8, i32 cpol)
__device__ __forceinline__ void tdm_load_1d(void* lds_dst, const void* gsrc,
                                            unsigned n_dwords) {
  unsigned lds_off = (unsigned)(unsigned long long)(uintptr_t)lds_dst;
  unsigned long long ga = (unsigned long long)(uintptr_t)gsrc;
  v4u g0;
  g0.x = 1u;                                        // count=1, user mode
  g0.y = lds_off;                                   // lds_addr [63:32]
  g0.z = (unsigned)(ga & 0xffffffffu);              // global_addr [95:64]
  g0.w = (unsigned)((ga >> 32) & 0x01ffffffu)       // global_addr [120:96]
       | (2u << 30);                                // type=2 ("image")
  v8i g1;
  g1[0] = (int)(2u << 16);                          // data_size=2 -> 4 bytes
  g1[1] = (int)((n_dwords & 0xffffu) << 16);        // tensor_dim0[15:0]
  g1[2] = (int)((n_dwords >> 16) & 0xffffu);        // tensor_dim0[31:16], dim1=0
  g1[3] = (int)((n_dwords & 0xffffu) << 16);        // tile_dim0 (<=65535)
  g1[4] = 0;                                        // tile_dim1/2 = 0 (unused)
  g1[5] = (int)n_dwords;                            // tensor_dim0_stride[31:0]
  g1[6] = 0;                                        // stride hi, dim1_stride lo
  g1[7] = 0;
  v4i gz4 = {0, 0, 0, 0};                           // groups 2/3: <=2-D tensor
  v8i gz8 = {0, 0, 0, 0, 0, 0, 0, 0};
  __builtin_amdgcn_tensor_load_to_lds(g0, g1, gz4, gz4, gz8, 0);
}

// ------------------------------------------------------- K0: f32->f16 swizzle
// dst[((ct*8+kc)*32 + l)*16 + e] = src[k*ldn + n],
//   n = ct*16 + (l&15),  k = kc*32 + 16*(l>>4) + e.
__global__ void cvt_swz_kernel(const float* __restrict__ src,
                               _Float16* __restrict__ dst, int ldn, int total) {
  int i = blockIdx.x * 256 + threadIdx.x;
  if (i >= total) return;
  int e  = i & 15;
  int l  = (i >> 4) & 31;
  int f  = i >> 9;
  int kc = f & 7;
  int ct = f >> 3;
  int n  = ct * 16 + (l & 15);
  int k  = kc * 32 + 16 * (l >> 4) + e;
  dst[i] = (_Float16)src[k * ldn + n];
}

// ---------------------------------------------------------------- K1: d2min
__global__ __launch_bounds__(256) void d2min_kernel(
    const float* __restrict__ dp, const float* __restrict__ pc,
    float* __restrict__ d2min) {
  __shared__ float tile[512 * 3];
  const int tid = threadIdx.x;
  const int gid = blockIdx.x * 256 + tid;          // 0..24575
  const int b   = gid / NDET;                      // block never straddles b
  const float qx = dp[gid * 3 + 0];
  const float qy = dp[gid * 3 + 1];
  const float qz = dp[gid * 3 + 2];
  float best = 3.0e38f;
  for (int t0 = 0; t0 < NPC; t0 += 512) {
    __syncthreads();
    if (tid < 32) {                                 // wave 0 drives the TDM
      tdm_load_1d(tile, pc + ((size_t)b * NPC + t0) * 3, 512 * 3);
      __builtin_amdgcn_s_wait_tensorcnt(0);
    }
    __syncthreads();
    #pragma unroll 4
    for (int l = 0; l < 512; ++l) {
      float dx = qx - tile[l * 3 + 0];
      float dy = qy - tile[l * 3 + 1];
      float dz = qz - tile[l * 3 + 2];
      best = fminf(best, dx * dx + dy * dy + dz * dz);
    }
  }
  d2min[gid] = best;
}

// ---------------------------------------------------------------- K2: partition
__global__ __launch_bounds__(256) void partition_kernel(
    const float* __restrict__ d2min,
    int* __restrict__ close_idx, int* __restrict__ far_idx) {
  __shared__ float sd[NDET];
  const int b = blockIdx.x / 48, chunk = blockIdx.x % 48, tid = threadIdx.x;
  for (int idx = tid; idx < NDET; idx += 256) sd[idx] = d2min[b * NDET + idx];
  __syncthreads();
  const int i  = chunk * 256 + tid;
  const float my = sd[i];
  int r = 0;
  for (int j = 0; j < NDET; ++j) {
    float v = sd[j];
    r += (v < my) | ((v == my) & (j < i));
  }
  if (r < MCLOSE) close_idx[b * MCLOSE + r] = i;
  else            far_idx[b * NFAR + (r - MCLOSE)] = i;
}

// ---------------------------------------------------------------- K3: knn+MLP
// LDS (dynamic, 66560 B -> 4 workgroups fit in a 320 KB WGP):
//   [0,4096)    persistent: q_xyz(192) q_gid(64) knn_idx(1536) knn_w(1536)
//   [4096,...)  phase1: pc_t 12288 | mer_d 24576 | mer_i 24576 | heads 1024
//               phase2 (reuse): Xswz 8192 | Yswz 8192 | Hf 8192 | bcs/b1s
template <int KNN>
__global__ __launch_bounds__(256) void knn_mlp_kernel(
    const float* __restrict__ pc, const float* __restrict__ dp,
    const float* __restrict__ feat,
    const _Float16* __restrict__ Wgs,  const float* __restrict__ bvec,
    const _Float16* __restrict__ W1gs, const float* __restrict__ b1vec,
    const float* __restrict__ W2,      const float* __restrict__ b2,
    const int* __restrict__ plist, int list_len,
    float* __restrict__ out) {
  extern __shared__ char smem_raw[];
  float* q_xyz   = (float*)(smem_raw);                 // [16][3]
  int*   q_gid   = (int*)  (smem_raw + 192);           // [16]
  int*   knn_idx = (int*)  (smem_raw + 256);           // [16][24]
  float* knn_w   = (float*)(smem_raw + 256 + 1536);    // [16][24]
  char*  ph      = smem_raw + 4096;
  // phase 1 views
  float* pc_t  = (float*)(ph);                         // [1024][3]
  float* mer_d = (float*)(ph + 12288);                 // [16][16][24]
  int*   mer_i = (int*)  (ph + 12288 + 24576);         // [16][16][24]
  int*   heads = (int*)  (ph + 12288 + 49152);         // [16][16]
  // phase 2 views (overlap phase 1 after barrier)
  _Float16* Xswz = (_Float16*)(ph);                    // [8][32][16]
  _Float16* Yswz = (_Float16*)(ph + 8192);             // [8][32][16]
  float*    Hf   = (float*)   (ph + 16384);            // [16][128]
  float*    bcs  = (float*)   (ph + 24576);            // [256]
  float*    b1s  = (float*)   (ph + 25600);            // [128]

  const int tid  = threadIdx.x;
  const int b    = blockIdx.y;
  const int grp  = blockIdx.x;
  const int lane = tid & 31;
  const int wave = tid >> 5;

  // ---- load the 16 query points of this group --------------------------
  if (tid < 16) {
    int g = plist[b * list_len + grp * 16 + tid];
    q_gid[tid] = g;
    q_xyz[tid * 3 + 0] = dp[((size_t)b * NDET + g) * 3 + 0];
    q_xyz[tid * 3 + 1] = dp[((size_t)b * NDET + g) * 3 + 1];
    q_xyz[tid * 3 + 2] = dp[((size_t)b * NDET + g) * 3 + 2];
  }
  __syncthreads();

  // ---- phase 1: top-KNN per query (16 sub-threads per query) -----------
  const int q = tid >> 4;          // 0..15 query
  const int s = tid & 15;          // 0..15 sub-scanner
  const float qx = q_xyz[q * 3 + 0], qy = q_xyz[q * 3 + 1], qz = q_xyz[q * 3 + 2];

  float bd[KNN]; int bi[KNN];
  #pragma unroll
  for (int t = 0; t < KNN; ++t) { bd[t] = 3.0e38f; bi[t] = 0; }

  for (int t0 = 0; t0 < NPC; t0 += 1024) {
    __syncthreads();
    if (tid < 32) {                                 // wave 0 drives the TDM
      tdm_load_1d(pc_t, pc + ((size_t)b * NPC + t0) * 3, 1024 * 3);
      __builtin_amdgcn_s_wait_tensorcnt(0);
    }
    __syncthreads();
    for (int l = s; l < 1024; l += 16) {
      float dx = qx - pc_t[l * 3 + 0];
      float dy = qy - pc_t[l * 3 + 1];
      float dz = qz - pc_t[l * 3 + 2];
      float d2 = dx * dx + dy * dy + dz * dz;
      if (d2 < bd[KNN - 1]) {                 // unrolled bubble insert
        float cd = d2; int ci = t0 + l;
        #pragma unroll
        for (int t = 0; t < KNN; ++t) {
          bool sw = cd < bd[t];
          float td = sw ? bd[t] : cd;  int ti = sw ? bi[t] : ci;
          bd[t] = sw ? cd : bd[t];     bi[t] = sw ? ci : bi[t];
          cd = td; ci = ti;
        }
      }
    }
  }
  __syncthreads();
  #pragma unroll
  for (int t = 0; t < KNN; ++t) {
    mer_d[(q * 16 + s) * 24 + t] = bd[t];
    mer_i[(q * 16 + s) * 24 + t] = bi[t];
  }
  __syncthreads();

  // ---- 16-way merge of sorted sub-lists + IDW weights (one thread/query)
  if (tid < 16) {
    const int mq = tid;
    for (int ss = 0; ss < 16; ++ss) heads[mq * 16 + ss] = 0;
    float wsum = 0.0f;
    for (int kk = 0; kk < KNN; ++kk) {
      float best = 3.0e38f; int bs = 0;
      for (int ss = 0; ss < 16; ++ss) {
        int hd = heads[mq * 16 + ss];
        float v = (hd < KNN) ? mer_d[(mq * 16 + ss) * 24 + hd] : 3.0e38f;
        if (v < best) { best = v; bs = ss; }
      }
      int hd = heads[mq * 16 + bs];
      knn_idx[mq * 24 + kk] = mer_i[(mq * 16 + bs) * 24 + hd];
      heads[mq * 16 + bs] = hd + 1;
      float w = 1.0f / (sqrtf(fmaxf(best, 1e-12f)) + 1e-8f);
      knn_w[mq * 24 + kk] = w;
      wsum += w;
    }
    float inv = 1.0f / wsum;
    for (int kk = 0; kk < KNN; ++kk) knn_w[mq * 24 + kk] *= inv;
  }
  __syncthreads();

  // ---- phase 2: biases + IDW feature gather, written A-fragment-swizzled
  bcs[tid] = bvec[tid];
  if (tid < 128) b1s[tid] = b1vec[tid];
  {
    const int kc  = tid >> 5;            // this thread's channel c = tid
    const int rem = tid & 31;
    int h, e; a_slot(rem, h, e);
    for (int qq = 0; qq < 16; ++qq) {
      float acc = 0.0f;
      for (int kk = 0; kk < KNN; ++kk) {
        int row = knn_idx[qq * 24 + kk];
        float w = knn_w[qq * 24 + kk];
        if (kk + 1 < KNN)   // hint next gather row toward L2/L0
          __builtin_prefetch(&feat[((size_t)b * NPC + knn_idx[qq * 24 + kk + 1]) * NCH + tid], 0, 1);
        acc += w * feat[((size_t)b * NPC + row) * NCH + tid];
      }
      Xswz[(kc * 32 + h * 16 + qq) * 16 + e] = (_Float16)acc;
    }
  }
  __syncthreads();

  // ---- layer 1: Y = relu(X @ W + b); A frags hoisted, B frags from L2 ---
  const int fr = lane & 15;        // frag row/col within tile
  const int fh = lane >> 4;        // lane-half select
  F16x16 afr[8];
  #pragma unroll
  for (int kc = 0; kc < 8; ++kc) {
    const v4u* pa = (const v4u*)&Xswz[(kc * 32 + lane) * 16];
    afr[kc].u[0] = pa[0]; afr[kc].u[1] = pa[1];
  }
  for (int ct = wave; ct < 16; ct += 8) {
    F32x8 acc;
    #pragma unroll
    for (int j = 0; j < 8; ++j) acc.e[j] = 0.0f;
    #pragma unroll
    for (int kc = 0; kc < 8; ++kc) {
      F16x16 bfr;
      const v4u* pb = (const v4u*)&Wgs[((ct * 8 + kc) * 32 + lane) * 16];
      bfr.u[0] = pb[0]; bfr.u[1] = pb[1];
      acc.v = __builtin_amdgcn_wmma_f32_16x16x32_f16(
          false, afr[kc].v, false, bfr.v, (short)0, acc.v, false, false);
    }
    #pragma unroll
    for (int j = 0; j < 8; ++j) {
      int m = j + 8 * fh, col = ct * 16 + fr;
      float v = fmaxf(acc.e[j] + bcs[col], 0.0f);
      int kc2 = col >> 5, rem2 = col & 31, h2, e2; a_slot(rem2, h2, e2);
      Yswz[(kc2 * 32 + h2 * 16 + m) * 16 + e2] = (_Float16)v;
    }
  }
  __syncthreads();

  // ---- layer 2: H = relu(Y @ W1 + b1)  (8 col tiles, one per wave) -----
  {
    const int ct = wave;
    F32x8 acc;
    #pragma unroll
    for (int j = 0; j < 8; ++j) acc.e[j] = 0.0f;
    #pragma unroll
    for (int kc = 0; kc < 8; ++kc) {
      F16x16 a2, b2f;
      const v4u* pa = (const v4u*)&Yswz[(kc * 32 + lane) * 16];
      a2.u[0] = pa[0]; a2.u[1] = pa[1];
      const v4u* pb = (const v4u*)&W1gs[((ct * 8 + kc) * 32 + lane) * 16];
      b2f.u[0] = pb[0]; b2f.u[1] = pb[1];
      acc.v = __builtin_amdgcn_wmma_f32_16x16x32_f16(
          false, a2.v, false, b2f.v, (short)0, acc.v, false, false);
    }
    #pragma unroll
    for (int j = 0; j < 8; ++j) {
      int m = j + 8 * fh, col = ct * 16 + fr;
      Hf[m * 128 + col] = fmaxf(acc.e[j] + b1s[col], 0.0f);
    }
  }
  __syncthreads();

  // ---- layer 3: tanh(H @ W2 + b2), keep 2 logits, scatter ---------------
  if (tid < 48) {
    int qq = tid / 3, o = tid % 3;
    float sacc = b2[o];
    #pragma unroll 4
    for (int j = 0; j < 128; ++j) sacc += Hf[qq * 128 + j] * W2[j * 3 + o];
    float t = tanhf(sacc);
    if (o < 2) out[((size_t)b * NDET + q_gid[qq]) * 2 + o] = t;
  }
}

// ---------------------------------------------------------------------------
extern "C" void kernel_launch(void* const* d_in, const int* in_sizes, int n_in,
                              void* d_out, int out_size, void* d_ws, size_t ws_size,
                              hipStream_t stream) {
  (void)in_sizes; (void)n_in; (void)out_size; (void)ws_size;
  const float* pc   = (const float*)d_in[0];   // (2,8192,3)
  const float* dp   = (const float*)d_in[1];   // (2,12288,3)
  const float* feat = (const float*)d_in[2];   // (2,8192,256)
  const float* Wc   = (const float*)d_in[3];   // (256,256)
  const float* bc   = (const float*)d_in[4];
  const float* W1c  = (const float*)d_in[5];   // (256,128)
  const float* b1c  = (const float*)d_in[6];
  const float* W2c  = (const float*)d_in[7];   // (128,3)
  const float* b2c  = (const float*)d_in[8];
  const float* Wf   = (const float*)d_in[9];
  const float* bf   = (const float*)d_in[10];
  const float* W1f  = (const float*)d_in[11];
  const float* b1f  = (const float*)d_in[12];
  const float* W2f  = (const float*)d_in[13];
  const float* b2f  = (const float*)d_in[14];
  float* out = (float*)d_out;                  // (2,12288,2)

  char* ws = (char*)d_ws;                      // needs ~576 KB scratch
  float*    d2min     = (float*)(ws + 0);           // 98304 B
  int*      close_idx = (int*)  (ws + 98304);       // 65536 B
  int*      far_idx   = (int*)  (ws + 163840);      // 32768 B
  _Float16* Wch       = (_Float16*)(ws + 196608);   // 131072 B (swizzled)
  _Float16* W1ch      = (_Float16*)(ws + 327680);   // 65536 B  (swizzled)
  _Float16* Wfh       = (_Float16*)(ws + 393216);   // 131072 B (swizzled)
  _Float16* W1fh      = (_Float16*)(ws + 524288);   // 65536 B  (swizzled)

  cvt_swz_kernel<<<(65536 + 255) / 256, 256, 0, stream>>>(Wc,  Wch,  256, 65536);
  cvt_swz_kernel<<<(32768 + 255) / 256, 256, 0, stream>>>(W1c, W1ch, 128, 32768);
  cvt_swz_kernel<<<(65536 + 255) / 256, 256, 0, stream>>>(Wf,  Wfh,  256, 65536);
  cvt_swz_kernel<<<(32768 + 255) / 256, 256, 0, stream>>>(W1f, W1fh, 128, 32768);

  d2min_kernel<<<96, 256, 0, stream>>>(dp, pc, d2min);
  partition_kernel<<<96, 256, 0, stream>>>(d2min, close_idx, far_idx);

  const size_t shmem = 66560;   // 4096 persistent + 62464 phase-1 (max phase)
  knn_mlp_kernel<24><<<dim3(MCLOSE / 16, NB), 256, shmem, stream>>>(
      pc, dp, feat, Wch, bc, W1ch, b1c, W2c, b2c, close_idx, MCLOSE, out);
  knn_mlp_kernel<12><<<dim3(NFAR / 16, NB), 256, shmem, stream>>>(
      pc, dp, feat, Wfh, bf, W1fh, b1f, W2f, b2f, far_idx, NFAR, out);
}
